// UniformMoERouter3_tucker_38165079392682
// MI455X (gfx1250) — compile-verified, matrix operand order checked by reading, not run on hardware
//
#include <hip/hip_runtime.h>
#include <hip/hip_bf16.h>

// ---------------------------------------------------------------------------
// Types for CDNA5 WMMA bf16 path
// ---------------------------------------------------------------------------
typedef __bf16 bf16;
typedef bf16  v16bf __attribute__((ext_vector_type(16)));
typedef bf16  v8bf  __attribute__((ext_vector_type(8)));
typedef bf16  v4bf  __attribute__((ext_vector_type(4)));
typedef float v8f   __attribute__((ext_vector_type(8)));

static __device__ __forceinline__ bf16 f2bf(float f) { return (bf16)f; }

#if __has_builtin(__builtin_amdgcn_sched_barrier)
#define SCHED_FENCE() __builtin_amdgcn_sched_barrier(0)
#else
#define SCHED_FENCE()
#endif

// ---------------------------------------------------------------------------
// Problem constants (from reference setup_inputs)
// ---------------------------------------------------------------------------
#define N_TOK   16384
#define D_IN    2048
#define D_H     1024
#define D_REL   512
#define D_GRAPH 512
#define D_Q     1024       // D_REL + D_GRAPH
#define NE      3
#define NBLK    (N_TOK / 256)   // 64 token-blocks for routing
#define TEMP_INV 10.0f     // 1 / 0.1
#define EPS_    1e-6f

// ---------------------------------------------------------------------------
// Elementwise fp32 -> bf16 (activations)
// ---------------------------------------------------------------------------
__global__ __launch_bounds__(256)
void cvt_f32_bf16(const float* __restrict__ in, bf16* __restrict__ out)
{
    const size_t i = (size_t)blockIdx.x * 256 + threadIdx.x;
    const float4 v = ((const float4*)in)[i];
    v4bf o = { f2bf(v.x), f2bf(v.y), f2bf(v.z), f2bf(v.w) };
    ((v4bf*)out)[i] = o;
}

// ---------------------------------------------------------------------------
// Weight transpose + convert:  B[K,N] fp32 row-major -> Bt[N,K] bf16 row-major
// 32x32 tiles through padded LDS; coalesced read along N, coalesced write
// along K (8-byte bf16x4 stores).
// ---------------------------------------------------------------------------
__global__ __launch_bounds__(256)
void transpose_cvt(const float* __restrict__ B, bf16* __restrict__ Bt, int K, int N)
{
    __shared__ float tile[32][33];
    const int kb = blockIdx.y * 32, nb = blockIdx.x * 32;
    {
        const int kk = threadIdx.x >> 3;
        const int nn = (threadIdx.x & 7) * 4;
        const float4 v = *(const float4*)&B[(size_t)(kb + kk) * N + nb + nn];
        tile[kk][nn + 0] = v.x; tile[kk][nn + 1] = v.y;
        tile[kk][nn + 2] = v.z; tile[kk][nn + 3] = v.w;
    }
    __syncthreads();
    {
        const int n  = threadIdx.x >> 3;
        const int k4 = (threadIdx.x & 7) * 4;
        v4bf o = { f2bf(tile[k4 + 0][n]), f2bf(tile[k4 + 1][n]),
                   f2bf(tile[k4 + 2][n]), f2bf(tile[k4 + 3][n]) };
        *(v4bf*)&Bt[(size_t)(nb + n) * K + kb + k4] = o;
    }
}

// ---------------------------------------------------------------------------
// Tiled WMMA GEMM (all-bf16 inputs):  C = relu(A * Bt^T + bias)
//   A : [M,K] bf16 row-major        Bt: [N,K] bf16 row-major
//   C : bf16 or fp32 (template), row stride ldc
// Block tile 128x128, K-step 32, double-buffered LDS (ping-pong, one barrier
// per iteration, branch-free steady state + peeled final tile). 8 waves; each
// wave 64x32 outputs = 4x2 fragments of v_wmma_f32_16x16x32_bf16.
// All dims are multiples of the tile sizes here.
// ---------------------------------------------------------------------------

// fragment build (ISA 16-bit A 16x32 lane/K layout; B mirrored lane=N) + 8 WMMA
#define COMPUTE_TILE(CUR)                                                        \
    {                                                                            \
        v16bf af[4];                                                             \
        _Pragma("unroll")                                                        \
        for (int mi = 0; mi < 4; ++mi) {                                         \
            const int r = wm + mi * 16 + ln;                                     \
            v8bf lo = *(const v8bf*)&Asl[CUR][r][kh];                            \
            v8bf hi = *(const v8bf*)&Asl[CUR][r][16 + kh];                       \
            af[mi] = __builtin_shufflevector(lo, hi,                             \
                0,1,2,3,4,5,6,7,8,9,10,11,12,13,14,15);                          \
        }                                                                        \
        v16bf bfm[2];                                                            \
        _Pragma("unroll")                                                        \
        for (int ni = 0; ni < 2; ++ni) {                                         \
            const int c = wn + ni * 16 + ln;                                     \
            v8bf lo = *(const v8bf*)&Bsl[CUR][c][kh];                            \
            v8bf hi = *(const v8bf*)&Bsl[CUR][c][16 + kh];                       \
            bfm[ni] = __builtin_shufflevector(lo, hi,                            \
                0,1,2,3,4,5,6,7,8,9,10,11,12,13,14,15);                          \
        }                                                                        \
        _Pragma("unroll")                                                        \
        for (int mi = 0; mi < 4; ++mi)                                           \
            _Pragma("unroll")                                                    \
            for (int ni = 0; ni < 2; ++ni)                                       \
                acc[mi][ni] = __builtin_amdgcn_wmma_f32_16x16x32_bf16(           \
                    false, af[mi], false, bfm[ni], (short)0, acc[mi][ni],        \
                    false, false);                                               \
    }

template<bool OUT_BF16>
__global__ __launch_bounds__(256)
void gemm_wmma(const bf16* __restrict__ A,
               const bf16* __restrict__ Bt,
               const float* __restrict__ bias,
               void* __restrict__ Cout,
               int M, int N, int K, int ldc)
{
    __shared__ __align__(16) bf16 Asl[2][128][32];
    __shared__ __align__(16) bf16 Bsl[2][128][32];

    const int tid  = threadIdx.x;
    const int lane = tid & 31;
    const int wave = tid >> 5;
    const int wm   = (wave >> 2) * 64;     // wave M offset (0 / 64)
    const int wn   = (wave & 3) * 32;      // wave N offset (0/32/64/96)

    const int mBase = blockIdx.y * 128;
    const int nBase = blockIdx.x * 128;

    const int ln = lane & 15;
    const int lh = lane >> 4;
    const int kh = lh * 8;                 // ISA 16-bit A 16x32 K-half select

    // tile loader mapping: 2x b128 chunks per matrix per thread
    const int r0 = tid >> 2;               // 0..63
    const int c0 = (tid & 3) * 8;          // 0,8,16,24

    // strength-reduced source pointers (advance by 32 elems per K-tile)
    const bf16* pa0 = A  + (size_t)(mBase + r0)      * K + c0;
    const bf16* pa1 = A  + (size_t)(mBase + r0 + 64) * K + c0;
    const bf16* pb0 = Bt + (size_t)(nBase + r0)      * K + c0;
    const bf16* pb1 = Bt + (size_t)(nBase + r0 + 64) * K + c0;

    v8f acc[4][2] = {};
    const int kTiles = K >> 5;

    // preload K-tile 0 into buffer 0
    v8bf ar0 = *(const v8bf*)pa0;
    v8bf ar1 = *(const v8bf*)pa1;
    v8bf br0 = *(const v8bf*)pb0;
    v8bf br1 = *(const v8bf*)pb1;
    *(v8bf*)&Asl[0][r0     ][c0] = ar0;
    *(v8bf*)&Asl[0][r0 + 64][c0] = ar1;
    *(v8bf*)&Bsl[0][r0     ][c0] = br0;
    *(v8bf*)&Bsl[0][r0 + 64][c0] = br1;
    __syncthreads();

    // branch-free steady state: every iteration has a next tile
    for (int kt = 0; kt < kTiles - 1; ++kt) {
        const int cur = kt & 1;
        const int nxt = cur ^ 1;

        pa0 += 32; pa1 += 32; pb0 += 32; pb1 += 32;
        ar0 = *(const v8bf*)pa0;           // issue next tile's loads first...
        ar1 = *(const v8bf*)pa1;
        br0 = *(const v8bf*)pb0;
        br1 = *(const v8bf*)pb1;
        SCHED_FENCE();                     // ...and keep them above the math

        COMPUTE_TILE(cur);                 // 12x ds_load_b128 + 8x wmma

        *(v8bf*)&Asl[nxt][r0     ][c0] = ar0;
        *(v8bf*)&Asl[nxt][r0 + 64][c0] = ar1;
        *(v8bf*)&Bsl[nxt][r0     ][c0] = br0;
        *(v8bf*)&Bsl[nxt][r0 + 64][c0] = br1;
        __syncthreads();
    }
    // peeled final tile (compute only)
    {
        const int cur = (kTiles - 1) & 1;
        COMPUTE_TILE(cur);
    }

    // epilogue: bias + relu; C/D layout: m = v + lh*8, n = ln
#pragma unroll
    for (int mi = 0; mi < 4; ++mi) {
#pragma unroll
        for (int ni = 0; ni < 2; ++ni) {
            const int gn = nBase + wn + ni * 16 + ln;
            const float bv = bias[gn];
#pragma unroll
            for (int v = 0; v < 8; ++v) {
                const int gm = mBase + wm + mi * 16 + v + lh * 8;
                float val = fmaxf(acc[mi][ni][v] + bv, 0.0f);
                if (OUT_BF16) ((bf16*)Cout)[(size_t)gm * ldc + gn] = f2bf(val);
                else          ((float*)Cout)[(size_t)gm * ldc + gn] = val;
            }
        }
    }
}

// ---------------------------------------------------------------------------
// Router: one wave32 per token. logits = q . embed_e / TEMP, softmax, argmax.
// Pure function of the token (no atomics -> deterministic).
// ---------------------------------------------------------------------------
__global__ __launch_bounds__(256)
void router_kernel(const float* __restrict__ q,
                   const float* __restrict__ embeds,
                   float* __restrict__ probs,
                   int*   __restrict__ choice)
{
    const int lane  = threadIdx.x & 31;
    const int token = blockIdx.x * 8 + (threadIdx.x >> 5);

    float s0 = 0.f, s1 = 0.f, s2 = 0.f;
    const float* qr = q + (size_t)token * D_Q;
    for (int k = lane * 4; k < D_Q; k += 128) {
        const float4 qv = *(const float4*)&qr[k];
        const float4 e0 = *(const float4*)&embeds[0 * D_Q + k];
        const float4 e1 = *(const float4*)&embeds[1 * D_Q + k];
        const float4 e2 = *(const float4*)&embeds[2 * D_Q + k];
        s0 += qv.x*e0.x + qv.y*e0.y + qv.z*e0.z + qv.w*e0.w;
        s1 += qv.x*e1.x + qv.y*e1.y + qv.z*e1.z + qv.w*e1.w;
        s2 += qv.x*e2.x + qv.y*e2.y + qv.z*e2.z + qv.w*e2.w;
    }
#pragma unroll
    for (int off = 16; off > 0; off >>= 1) {
        s0 += __shfl_xor(s0, off, 32);
        s1 += __shfl_xor(s1, off, 32);
        s2 += __shfl_xor(s2, off, 32);
    }

    if (lane == 0) {
        float l[NE] = { s0 * TEMP_INV, s1 * TEMP_INV, s2 * TEMP_INV };
        const float m = fmaxf(l[0], fmaxf(l[1], l[2]));
        float e[NE]; float sum = 0.f;
#pragma unroll
        for (int i = 0; i < NE; ++i) { e[i] = __expf(l[i] - m); sum += e[i]; }
        const float inv = 1.0f / sum;
        int best = 0; float bp = -1.f;
#pragma unroll
        for (int i = 0; i < NE; ++i) {
            const float p = e[i] * inv;
            probs[(size_t)token * NE + i] = p;
            if (p > bp) { bp = p; best = i; }
        }
        choice[token] = best;
    }
}

// ---------------------------------------------------------------------------
// Per-block expert histogram (deterministic routing, step 1)
// ---------------------------------------------------------------------------
__global__ __launch_bounds__(256)
void hist_kernel(const int* __restrict__ choice, int* __restrict__ blockHist)
{
    __shared__ int h[NE];
    if (threadIdx.x < NE) h[threadIdx.x] = 0;
    __syncthreads();
    atomicAdd(&h[choice[blockIdx.x * 256 + threadIdx.x]], 1);   // block-local only
    __syncthreads();
    if (threadIdx.x < NE) blockHist[blockIdx.x * NE + threadIdx.x] = h[threadIdx.x];
}

// ---------------------------------------------------------------------------
// Deterministic fixed-order reduction of importance[e] and gate_sum[e]
// ---------------------------------------------------------------------------
__global__ __launch_bounds__(256)
void reduce_kernel(const float* __restrict__ probs, const int* __restrict__ choice,
                   float* __restrict__ importance, float* __restrict__ gate_sum)
{
    float imp[NE] = {0.f, 0.f, 0.f};
    float gs[NE]  = {0.f, 0.f, 0.f};
    for (int t = threadIdx.x; t < N_TOK; t += 256) {
        const int e = choice[t];
#pragma unroll
        for (int i = 0; i < NE; ++i) {
            const float p = probs[(size_t)t * NE + i];
            imp[i] += p;
            if (i == e) gs[i] += p;
        }
    }
    __shared__ float red[256];
#pragma unroll
    for (int i = 0; i < NE; ++i) {
        red[threadIdx.x] = imp[i];
        __syncthreads();
        for (int s = 128; s > 0; s >>= 1) {
            if (threadIdx.x < s) red[threadIdx.x] += red[threadIdx.x + s];
            __syncthreads();
        }
        if (threadIdx.x == 0) importance[i] = red[0];
        __syncthreads();
        red[threadIdx.x] = gs[i];
        __syncthreads();
        for (int s = 128; s > 0; s >>= 1) {
            if (threadIdx.x < s) red[threadIdx.x] += red[threadIdx.x + s];
            __syncthreads();
        }
        if (threadIdx.x == 0) gate_sum[i] = red[0];
        __syncthreads();
    }
}

// ---------------------------------------------------------------------------
// Scan block histograms -> per-block bases; emit counts + loss.
//   distribution_loss = sum_e counts_e * (gate_sum_e/counts_e) / E = sum gate/E
//   cv = std(importance, ddof=1) / (mean + eps); loss = dist + max(cv, 0.1)
// ---------------------------------------------------------------------------
__global__ void scan_finalize_kernel(const int* __restrict__ blockHist,
                                     const float* __restrict__ gate_sum,
                                     const float* __restrict__ importance,
                                     int* __restrict__ blockBase,
                                     float* __restrict__ out_loss,
                                     float* __restrict__ out_counts)
{
    if (threadIdx.x != 0 || blockIdx.x != 0) return;
    int counts[NE] = {0, 0, 0};
    for (int b = 0; b < NBLK; ++b)
        for (int e = 0; e < NE; ++e) counts[e] += blockHist[b * NE + e];
    int run[NE];
    run[0] = 0; run[1] = counts[0]; run[2] = counts[0] + counts[1];
    for (int b = 0; b < NBLK; ++b)
        for (int e = 0; e < NE; ++e) {
            blockBase[b * NE + e] = run[e];
            run[e] += blockHist[b * NE + e];
        }
    for (int e = 0; e < NE; ++e) out_counts[e] = (float)counts[e];

    const float dist = (gate_sum[0] + gate_sum[1] + gate_sum[2]) / (float)NE;
    const float i0 = importance[0], i1 = importance[1], i2 = importance[2];
    const float mean = (i0 + i1 + i2) / 3.0f;
    const float d0 = i0 - mean, d1 = i1 - mean, d2 = i2 - mean;
    const float var = (d0*d0 + d1*d1 + d2*d2) / 2.0f;       // ddof = 1
    const float cv  = __fsqrt_rn(var) / (mean + EPS_);
    out_loss[0] = dist + fmaxf(cv, 0.1f);
}

// ---------------------------------------------------------------------------
// Stable placement via wave ballots + per-wave LDS scan (deterministic order:
// tokens within an expert keep ascending index order).
// ---------------------------------------------------------------------------
__global__ __launch_bounds__(256)
void place_kernel(const int* __restrict__ choice,
                  const int* __restrict__ blockBase,
                  int* __restrict__ all_idx_i,
                  float* __restrict__ out_all_idx)
{
    __shared__ int wbase[8][NE];
    const int tid  = threadIdx.x;
    const int lane = tid & 31;
    const int wave = tid >> 5;
    const int t = blockIdx.x * 256 + tid;
    const int e = choice[t];

    int lower = 0;
#pragma unroll
    for (int e2 = 0; e2 < NE; ++e2) {
        const unsigned mm = (unsigned)__ballot(e == e2);     // wave32: low 32 bits
        if (e2 == e) lower = __popc(mm & ((1u << lane) - 1u));
        if (lane == 0) wbase[wave][e2] = __popc(mm);
    }
    __syncthreads();
    if (tid < NE) {                       // exclusive scan over 8 waves, expert tid
        int s = 0;
        for (int w = 0; w < 8; ++w) { const int c = wbase[w][tid]; wbase[w][tid] = s; s += c; }
    }
    __syncthreads();

    const int pos = blockBase[blockIdx.x * NE + e] + wbase[wave][e] + lower;
    all_idx_i[pos] = t;
    out_all_idx[pos] = (float)t;
}

// ---------------------------------------------------------------------------
// Gather expert_data = x[all_idx]  (one block per output row, float4 copies)
// ---------------------------------------------------------------------------
__global__ __launch_bounds__(256)
void gather_kernel(const float* __restrict__ x,
                   const int* __restrict__ all_idx_i,
                   float* __restrict__ out)
{
    const int p = blockIdx.x;
    const int t = all_idx_i[p];
    const float4* src = (const float4*)(x + (size_t)t * D_IN);
    float4* dst = (float4*)(out + (size_t)p * D_IN);
#pragma unroll
    for (int i = 0; i < 2; ++i) {
        const int j = threadIdx.x + i * 256;   // 512 float4 per row
        dst[j] = src[j];
    }
}

// ---------------------------------------------------------------------------
extern "C" void kernel_launch(void* const* d_in, const int* in_sizes, int n_in,
                              void* d_out, int out_size, void* d_ws, size_t ws_size,
                              hipStream_t stream) {
    const float* x      = (const float*)d_in[0];
    const float* y      = (const float*)d_in[1];
    const float* w1_rel = (const float*)d_in[2];
    const float* b1_rel = (const float*)d_in[3];
    const float* w2_rel = (const float*)d_in[4];
    const float* b2_rel = (const float*)d_in[5];
    const float* w1_gra = (const float*)d_in[6];
    const float* b1_gra = (const float*)d_in[7];
    const float* w2_gra = (const float*)d_in[8];
    const float* b2_gra = (const float*)d_in[9];
    const float* embeds = (const float*)d_in[10];

    float* out = (float*)d_out;
    float* out_expert  = out;                                  // [N_TOK * D_IN]
    float* out_loss    = out + (size_t)N_TOK * D_IN;           // [1]
    float* out_all_idx = out_loss + 1;                         // [N_TOK]
    float* out_counts  = out_all_idx + N_TOK;                  // [NE]

    // scratch layout (bf16 region first, then fp32/int region)
    bf16* Ab  = (bf16*)d_ws;                                   // [N_TOK * D_IN]
    bf16* Hb  = Ab  + (size_t)N_TOK * D_IN;                    // [N_TOK * D_H]
    bf16* W1t = Hb  + (size_t)N_TOK * D_H;                     // [D_H  * D_IN]
    bf16* W2t = W1t + (size_t)D_H * D_IN;                      // [D_REL* D_H]
    float* q     = (float*)(W2t + (size_t)D_REL * D_H);        // [N_TOK * D_Q]
    float* probs = q + (size_t)N_TOK * D_Q;                    // [N_TOK * NE]
    float* importance = probs + (size_t)N_TOK * NE;            // [4]
    float* gate_sum   = importance + 4;                        // [4]
    int* choice     = (int*)(gate_sum + 4);                    // [N_TOK]
    int* all_idx_i  = choice + N_TOK;                          // [N_TOK]
    int* blockHist  = all_idx_i + N_TOK;                       // [NBLK * NE]
    int* blockBase  = blockHist + NBLK * NE;                   // [NBLK * NE]

    const dim3 blk(256);

    // ---- MLP 1 (rel): Hb = relu(x @ w1_rel + b1); q[:,:512] = relu(Hb @ w2_rel + b2)
    cvt_f32_bf16<<<(N_TOK * D_IN / 4) / 256, blk, 0, stream>>>(x, Ab);
    transpose_cvt<<<dim3(D_H / 32, D_IN / 32), blk, 0, stream>>>(w1_rel, W1t, D_IN, D_H);
    gemm_wmma<true><<<dim3(D_H / 128, N_TOK / 128), blk, 0, stream>>>(
        Ab, W1t, b1_rel, Hb, N_TOK, D_H, D_IN, D_H);
    transpose_cvt<<<dim3(D_REL / 32, D_H / 32), blk, 0, stream>>>(w2_rel, W2t, D_H, D_REL);
    gemm_wmma<false><<<dim3(D_REL / 128, N_TOK / 128), blk, 0, stream>>>(
        Hb, W2t, b2_rel, q, N_TOK, D_REL, D_H, D_Q);

    // ---- MLP 2 (graph): reuse Ab/Hb/W1t/W2t; q[:,512:] = ...
    cvt_f32_bf16<<<(N_TOK * D_IN / 4) / 256, blk, 0, stream>>>(y, Ab);
    transpose_cvt<<<dim3(D_H / 32, D_IN / 32), blk, 0, stream>>>(w1_gra, W1t, D_IN, D_H);
    gemm_wmma<true><<<dim3(D_H / 128, N_TOK / 128), blk, 0, stream>>>(
        Ab, W1t, b1_gra, Hb, N_TOK, D_H, D_IN, D_H);
    transpose_cvt<<<dim3(D_GRAPH / 32, D_H / 32), blk, 0, stream>>>(w2_gra, W2t, D_H, D_GRAPH);
    gemm_wmma<false><<<dim3(D_GRAPH / 128, N_TOK / 128), blk, 0, stream>>>(
        Hb, W2t, b2_gra, q + D_REL, N_TOK, D_GRAPH, D_H, D_Q);

    // ---- Routing (fully deterministic: no global atomics anywhere)
    router_kernel<<<N_TOK / 8, blk, 0, stream>>>(q, embeds, probs, choice);
    hist_kernel<<<NBLK, blk, 0, stream>>>(choice, blockHist);
    reduce_kernel<<<1, blk, 0, stream>>>(probs, choice, importance, gate_sum);
    scan_finalize_kernel<<<1, 1, 0, stream>>>(blockHist, gate_sum, importance,
                                              blockBase, out_loss, out_counts);
    place_kernel<<<NBLK, blk, 0, stream>>>(choice, blockBase, all_idx_i, out_all_idx);
    gather_kernel<<<N_TOK, blk, 0, stream>>>(x, all_idx_i, out_expert);
}
